// EncoderBlock_14697378087355
// MI455X (gfx1250) — matrix operand order repaired and checked by reading, not compiled, not run on hardware
//
#include <hip/hip_runtime.h>

// ---------------------------------------------------------------- constants
#define BB 16
#define TT 1024
#define CC 256
#define HH 8
#define DD 32
#define FF 1024
#define BT (BB*TT)

typedef __bf16 bf16;
typedef __attribute__((ext_vector_type(16))) __bf16 v16bf;
typedef __attribute__((ext_vector_type(8)))  float  v8f;
typedef unsigned short u16;
typedef __attribute__((ext_vector_type(8))) u16 u16x8;
typedef __attribute__((ext_vector_type(4))) u16 u16x4;

union Frag16 { v16bf v; u16x8 h[2]; };
union VRow   { u16x8 h[4]; __bf16 e[32]; };
union Pack8  { u16x8 h;    __bf16 e[8];  };
union Pack4  { u16x4 h;    __bf16 e[4];  };

__device__ __forceinline__ v8f wmma_bf16(v16bf a, v16bf b, v8f c) {
  // D = A(16x32 bf16) * B(32x16 bf16) + C(16x16 f32)
  return __builtin_amdgcn_wmma_f32_16x16x32_bf16(false, a, false, b, (short)0, c,
                                                 false, false);
}

// gfx1250 async global->LDS copy (ISA §15.18.3, GV mode), tracked by ASYNCcnt.
__device__ __forceinline__ void async_copy_b128(unsigned lds_byte_off,
                                                const void* gaddr) {
  asm volatile("global_load_async_to_lds_b128 %0, %1, off"
               :: "v"(lds_byte_off), "v"((unsigned long long)gaddr)
               : "memory");
}
__device__ __forceinline__ void wait_asynccnt0() {
  asm volatile("s_wait_asynccnt 0" ::: "memory");
}

// ---------------------------------------------------------------- layernorm
// One wave per 256-element row; 8 rows per 256-thread block. Output bf16.
__global__ __launch_bounds__(256) void ln_bf16_kernel(
    const float* __restrict__ x, const float* __restrict__ g,
    const float* __restrict__ bb, bf16* __restrict__ out)
{
  int lane = threadIdx.x & 31;
  int row  = blockIdx.x * 8 + (threadIdx.x >> 5);
  const float* xr = x + (size_t)row * CC + lane * 8;
  float4 a = *(const float4*)xr;
  float4 b = *(const float4*)(xr + 4);
  float vals[8] = {a.x, a.y, a.z, a.w, b.x, b.y, b.z, b.w};
  float s = 0.f, sq = 0.f;
  #pragma unroll
  for (int i = 0; i < 8; ++i) { s += vals[i]; sq += vals[i] * vals[i]; }
  #pragma unroll
  for (int mk = 1; mk < 32; mk <<= 1) {
    s  += __shfl_xor(s,  mk, 32);
    sq += __shfl_xor(sq, mk, 32);
  }
  float mu  = s * (1.0f / CC);
  float var = sq * (1.0f / CC) - mu * mu;
  float rs  = rsqrtf(var + 1e-5f);
  const float* gp = g  + lane * 8;
  const float* bp = bb + lane * 8;
  Pack8 o;
  #pragma unroll
  for (int i = 0; i < 8; ++i)
    o.e[i] = (__bf16)((vals[i] - mu) * rs * gp[i] + bp[i]);
  *(u16x8*)(out + (size_t)row * CC + lane * 8) = o.h;
}

// ---------------------------------------------------------- weight prep
// WqkvT[768][256] bf16 : row n = {sel,h,d}; WqkvT[n][c] = w{q,k,v}[h][c][d]
__global__ void prep_qkvT_kernel(const float* __restrict__ wq,
                                 const float* __restrict__ wk,
                                 const float* __restrict__ wv,
                                 bf16* __restrict__ out)
{
  int idx = blockIdx.x * blockDim.x + threadIdx.x;   // 768*256 threads
  int n = idx >> 8, c = idx & 255;
  int sel = n >> 8, h = (n >> 5) & 7, d = n & 31;
  const float* w = (sel == 0) ? wq : ((sel == 1) ? wk : wv);
  out[idx] = (bf16)w[((h << 8) + c) * 32 + d];
}

// Transpose f32 src[R][Cc] -> bf16 dst[Cc][R], 32x32 tiles staged through LDS
// with gfx1250 async global->LDS copies (ASYNCcnt).
__global__ __launch_bounds__(256) void transpose_async_kernel(
    const float* __restrict__ src, bf16* __restrict__ dst, int R, int Cc)
{
  __shared__ __align__(16) float tile[32 * 36];   // 36-f stride: 16B-aligned rows
  int t  = threadIdx.x;
  int rt0 = blockIdx.x * 32;   // src row base
  int ct0 = blockIdx.y * 32;   // src col base

  // stage: thread t copies 16B: src row (rt0 + t/8), cols ct0 + (t%8)*4 ..+3
  {
    int r  = t >> 3, cq = t & 7;
    const float* gp = src + (size_t)(rt0 + r) * Cc + ct0 + cq * 4;
    unsigned lds_off = (unsigned)(size_t)&tile[r * 36 + cq * 4];
    async_copy_b128(lds_off, gp);
  }
  wait_asynccnt0();
  __syncthreads();

  // drain: thread t writes dst column c2 = t/8, src rows rt0 + (t%8)*4 ..+3
  {
    int c2 = t >> 3, rq = (t & 7) * 4;
    Pack4 o;
    #pragma unroll
    for (int i = 0; i < 4; ++i)
      o.e[i] = (__bf16)tile[(rq + i) * 36 + c2];
    *(u16x4*)(dst + (size_t)(ct0 + c2) * R + rt0 + rq) = o.h;
  }
}

// ---------------------------------------------------------------- GEMM
// C[M][N] = A[M][K](bf16,row-major) * Bt[N][K](bf16)  + bias + res, opt ReLU.
// 8 waves/block; wave tile 16(M) x 64(N); block tile 32 x 256.
// K-loop software-pipelined (double-buffered fragments), fully branchless
// body so the scheduler can hoist loads a whole stage ahead. K % 64 == 0.
__device__ __forceinline__ Frag16 load_frag_a(const bf16* ap, int kk, int kb) {
  Frag16 a;
  a.h[0] = *(const u16x8*)(ap + kk + kb);
  a.h[1] = *(const u16x8*)(ap + kk + kb + 8);
  return a;
}
__device__ __forceinline__ Frag16 load_frag_b(const bf16* bp) {
  Frag16 b;
  b.h[0] = *(const u16x8*)bp;
  b.h[1] = *(const u16x8*)(bp + 8);
  return b;
}

__global__ __launch_bounds__(256) void gemm_bf16_kernel(
    const bf16* __restrict__ A, const bf16* __restrict__ Bt,
    const float* __restrict__ bias, const float* __restrict__ res,
    float* __restrict__ outf, bf16* __restrict__ outb,
    int M, int N, int K, int relu)
{
  int tid = threadIdx.x, lane = tid & 31, wid = tid >> 5;
  int wm = wid >> 2, wn = wid & 3;
  int m0 = blockIdx.x * 32 + wm * 16;
  int n0 = blockIdx.y * 256 + wn * 64;
  int lm = lane & 15, lh = lane >> 4, kb = lh * 16;

  const bf16* ap = A + (size_t)(m0 + lm) * K;
  const bf16* bp[4];
  #pragma unroll
  for (int j = 0; j < 4; ++j)
    bp[j] = Bt + (size_t)(n0 + j * 16 + lm) * K + lh * 16;

  v8f acc[4] = {v8f{}, v8f{}, v8f{}, v8f{}};

  // pipeline stage 0
  Frag16 a0 = load_frag_a(ap, 0, kb);
  Frag16 b0[4];
  #pragma unroll
  for (int j = 0; j < 4; ++j) b0[j] = load_frag_b(bp[j]);

  for (int kk = 0; kk < K; kk += 64) {
    // issue stage (kk+32) loads while stage kk computes
    Frag16 a1 = load_frag_a(ap, kk + 32, kb);
    Frag16 b1[4];
    #pragma unroll
    for (int j = 0; j < 4; ++j) b1[j] = load_frag_b(bp[j] + kk + 32);

    // branchless prefetch of the A stream two stages ahead
    int kpf = (kk + 128 < K) ? kk + 128 : 0;
    __builtin_prefetch(ap + kpf, 0, 3);

    #pragma unroll
    for (int j = 0; j < 4; ++j) acc[j] = wmma_bf16(a0.v, b0[j].v, acc[j]);

    // issue stage (kk+64) loads (clamped, branchless) while kk+32 computes
    int kn = (kk + 64 < K) ? kk + 64 : 0;
    a0 = load_frag_a(ap, kn, kb);
    #pragma unroll
    for (int j = 0; j < 4; ++j) b0[j] = load_frag_b(bp[j] + kn);

    #pragma unroll
    for (int j = 0; j < 4; ++j) acc[j] = wmma_bf16(a1.v, b1[j].v, acc[j]);
  }

  #pragma unroll
  for (int j = 0; j < 4; ++j) {
    #pragma unroll
    for (int r = 0; r < 8; ++r) {
      int row = m0 + 8 * lh + r;
      int col = n0 + j * 16 + lm;
      float v = acc[j][r];
      if (bias) v += bias[col];
      if (res)  v += res[(size_t)row * N + col];
      if (relu) v = fmaxf(v, 0.f);
      size_t o = (size_t)row * N + col;
      if (outf) outf[o] = v;
      if (outb) outb[o] = (bf16)v;
    }
  }
}

// ------------------------------------------------------------- attention
// qkv[bt][768]: cols 0-255 = Q (h-major, d inner), 256-511 = K, 512-767 = V.
// One wave per (b, h, 16-row Q tile). Flash-style online softmax.
__global__ __launch_bounds__(32) void attn_kernel(const bf16* __restrict__ qkv,
                                                  bf16* __restrict__ attb)
{
  __shared__ __align__(16) __bf16 Pl[16 * 32];   // P tile, row-major 16x32
  __shared__ __align__(16) __bf16 Vt[32 * 40];   // V^T tile [d][s], padded

  int lane = threadIdx.x;
  int rt = blockIdx.x, h = blockIdx.y, b = blockIdx.z;
  int lm = lane & 15, lh = lane >> 4, kb = lh * 16;

  const bf16* base = qkv + (size_t)b * TT * 768;
  const bf16* Qp = base + h * 32;
  const bf16* Kp = base + 256 + h * 32;
  const bf16* Vp = base + 512 + h * 32;

  // Q fragment (A-matrix 16x32): lane lm = row, K-halves by lane[4]
  Frag16 qf;
  {
    const bf16* qr = Qp + (size_t)(rt * 16 + lm) * 768;
    qf.h[0] = *(const u16x8*)(qr + kb);
    qf.h[1] = *(const u16x8*)(qr + kb + 8);
  }

  float mrow[8], lrow[8];
  #pragma unroll
  for (int r = 0; r < 8; ++r) { mrow[r] = -1e30f; lrow[r] = 0.f; }
  v8f O0 = {}, O1 = {};

  int jmax = (rt * 16 + 15) >> 5;
  for (int jt = 0; jt <= jmax; ++jt) {
    // --- S = Q * K^T : two 16-col subtiles. B-frag is K rows directly.
    Frag16 k0f, k1f;
    {
      const bf16* kr = Kp + (size_t)(jt * 32 + lm) * 768 + lh * 16;
      k0f.h[0] = *(const u16x8*)kr; k0f.h[1] = *(const u16x8*)(kr + 8);
    }
    {
      const bf16* kr = Kp + (size_t)(jt * 32 + 16 + lm) * 768 + lh * 16;
      k1f.h[0] = *(const u16x8*)kr; k1f.h[1] = *(const u16x8*)(kr + 8);
    }
    v8f z = {};
    v8f S0 = wmma_bf16(qf.v, k0f.v, z);
    v8f S1 = wmma_bf16(qf.v, k1f.v, z);

    // --- stage V chunk transposed into LDS: Vt[d][s]
    {
      const bf16* vrp = Vp + (size_t)(jt * 32 + lane) * 768;
      VRow vr;
      vr.h[0] = *(const u16x8*)(vrp + 0);
      vr.h[1] = *(const u16x8*)(vrp + 8);
      vr.h[2] = *(const u16x8*)(vrp + 16);
      vr.h[3] = *(const u16x8*)(vrp + 24);
      #pragma unroll
      for (int d = 0; d < 32; ++d) Vt[d * 40 + lane] = vr.e[d];
    }

    // --- scale, causal mask, online softmax (C-layout: lane=col, vgpr=row)
    float p0[8], p1[8];
    #pragma unroll
    for (int r = 0; r < 8; ++r) {
      int mg = rt * 16 + 8 * lh + r;
      int s0 = jt * 32 + lm;
      float a0 = S0[r] * 0.0625f; if (s0      > mg) a0 = -1e30f;
      float a1 = S1[r] * 0.0625f; if (s0 + 16 > mg) a1 = -1e30f;
      float cm = fmaxf(a0, a1);
      #pragma unroll
      for (int mk = 1; mk < 16; mk <<= 1)
        cm = fmaxf(cm, __shfl_xor(cm, mk, 32));
      float nm    = fmaxf(mrow[r], cm);
      float alpha = __expf(mrow[r] - nm);
      float e0 = __expf(a0 - nm);
      float e1 = __expf(a1 - nm);
      float cs = e0 + e1;
      #pragma unroll
      for (int mk = 1; mk < 16; mk <<= 1)
        cs += __shfl_xor(cs, mk, 32);
      lrow[r] = lrow[r] * alpha + cs;
      mrow[r] = nm;
      O0[r] *= alpha;
      O1[r] *= alpha;
      p0[r] = e0; p1[r] = e1;
    }

    // --- P (C-layout) -> LDS row-major -> A-fragment
    #pragma unroll
    for (int r = 0; r < 8; ++r) {
      int m = 8 * lh + r;
      Pl[m * 32 + lm]      = (__bf16)p0[r];
      Pl[m * 32 + 16 + lm] = (__bf16)p1[r];
    }
    __syncthreads();   // single-wave WG: acts as compiler/LDS fence

    Frag16 pf, v0f, v1f;
    pf.h[0] = *(const u16x8*)(&Pl[lm * 32 + kb]);
    pf.h[1] = *(const u16x8*)(&Pl[lm * 32 + kb + 8]);
    {
      const __bf16* vtp = &Vt[(size_t)lm * 40 + lh * 16];
      v0f.h[0] = *(const u16x8*)vtp; v0f.h[1] = *(const u16x8*)(vtp + 8);
    }
    {
      const __bf16* vtp = &Vt[(size_t)(16 + lm) * 40 + lh * 16];
      v1f.h[0] = *(const u16x8*)vtp; v1f.h[1] = *(const u16x8*)(vtp + 8);
    }
    O0 = wmma_bf16(pf.v, v0f.v, O0);
    O1 = wmma_bf16(pf.v, v1f.v, O1);
    __syncthreads();
  }

  // --- finalize: O /= l, write bf16 att output [bt][h*32+d]
  #pragma unroll
  for (int r = 0; r < 8; ++r) {
    float inv = 1.0f / lrow[r];
    int m = 8 * lh + r;
    size_t row = (size_t)(b * TT + rt * 16 + m) * 256 + h * 32;
    attb[row + lm]      = (bf16)(O0[r] * inv);
    attb[row + 16 + lm] = (bf16)(O1[r] * inv);
  }
}

// ---------------------------------------------------------------- launch
extern "C" void kernel_launch(void* const* d_in, const int* in_sizes, int n_in,
                              void* d_out, int out_size, void* d_ws, size_t ws_size,
                              hipStream_t stream)
{
  const float* x      = (const float*)d_in[0];
  const float* wq     = (const float*)d_in[1];
  const float* wk     = (const float*)d_in[2];
  const float* wv     = (const float*)d_in[3];
  const float* w_proj = (const float*)d_in[4];
  const float* b_proj = (const float*)d_in[5];
  const float* w1     = (const float*)d_in[6];
  const float* b1     = (const float*)d_in[7];
  const float* w2     = (const float*)d_in[8];
  const float* b2     = (const float*)d_in[9];
  const float* ln1_g  = (const float*)d_in[10];
  const float* ln1_b  = (const float*)d_in[11];
  const float* ln2_g  = (const float*)d_in[12];
  const float* ln2_b  = (const float*)d_in[13];

  char* w = (char*)d_ws;
  size_t off = 0;
  auto carve = [&](size_t bytes) -> char* {
    char* p = w + off;
    off = (off + bytes + 255) & ~(size_t)255;
    return p;
  };
  bf16*  hbf   = (bf16*)carve((size_t)BT * CC * 2);       // ln1(x) bf16
  bf16*  wqkvT = (bf16*)carve((size_t)768 * CC * 2);      // fused QKV weight^T
  bf16*  wpT   = (bf16*)carve((size_t)CC * CC * 2);
  bf16*  w1T   = (bf16*)carve((size_t)FF * CC * 2);
  bf16*  w2T   = (bf16*)carve((size_t)CC * FF * 2);
  bf16*  qkvb  = (bf16*)carve((size_t)BT * 768 * 2);      // Q|K|V bf16
  bf16*  attb  = (bf16*)carve((size_t)BT * CC * 2);       // attention out bf16
  float* x2    = (float*)carve((size_t)BT * CC * 4);      // x + attn proj (f32)
  bf16*  h2b   = (bf16*)carve((size_t)BT * CC * 2);       // ln2(x2) bf16
  bf16*  ff1b  = (bf16*)carve((size_t)BT * FF * 2);       // relu(h2@w1+b1) bf16
  (void)ws_size; (void)in_sizes; (void)n_in; (void)out_size;

  // 0) weights -> bf16 transposed layouts (async-LDS staged transposes)
  prep_qkvT_kernel<<<(768 * 256) / 256, 256, 0, stream>>>(wq, wk, wv, wqkvT);
  transpose_async_kernel<<<dim3(8, 8),  256, 0, stream>>>(w_proj, wpT, 256, 256);
  transpose_async_kernel<<<dim3(8, 32), 256, 0, stream>>>(w1,     w1T, 256, 1024);
  transpose_async_kernel<<<dim3(32, 8), 256, 0, stream>>>(w2,     w2T, 1024, 256);

  // 1) h = ln1(x)  (bf16)
  ln_bf16_kernel<<<BT / 8, 256, 0, stream>>>(x, ln1_g, ln1_b, hbf);

  // 2) QKV = h @ WqkvT   (M=16384, N=768, K=256)
  gemm_bf16_kernel<<<dim3(BT / 32, 3), 256, 0, stream>>>(
      hbf, wqkvT, nullptr, nullptr, nullptr, qkvb, BT, 768, 256, 0);

  // 3) causal flash attention per (b, h, 16-row tile)
  attn_kernel<<<dim3(TT / 16, HH, BB), 32, 0, stream>>>(qkvb, attb);

  // 4) x2 = x + att @ w_proj + b_proj   (f32)
  gemm_bf16_kernel<<<dim3(BT / 32, 1), 256, 0, stream>>>(
      attb, wpT, b_proj, x, x2, nullptr, BT, 256, 256, 0);

  // 5) h2 = ln2(x2)  (bf16)
  ln_bf16_kernel<<<BT / 8, 256, 0, stream>>>(x2, ln2_g, ln2_b, h2b);

  // 6) f1 = relu(h2 @ w1 + b1)  (M=16384, N=1024, K=256) -> bf16
  gemm_bf16_kernel<<<dim3(BT / 32, 4), 256, 0, stream>>>(
      h2b, w1T, b1, nullptr, nullptr, ff1b, BT, 1024, 256, 1);

  // 7) out = x2 + f1 @ w2 + b2  (M=16384, N=256, K=1024) -> f32 d_out
  gemm_bf16_kernel<<<dim3(BT / 32, 1), 256, 0, stream>>>(
      ff1b, w2T, b2, x2, (float*)d_out, nullptr, BT, 256, 1024, 0);
}
